// LDSModel_89292370084272
// MI455X (gfx1250) — compile-verified, matrix operand order checked by reading, not run on hardware
//
#include <hip/hip_runtime.h>

// ---------------------------------------------------------------------------
// MI455X (gfx1250, wave32) implementation of the FOH linear-ODE reference:
//   Eyy = exp(A h); Eyu = h*phi1(Ah)B; Eyv = h^2*phi2(Ah)B   (Taylor, ||Ah||~1e-2)
//   c[t] = G0 u[t] + G1 u[t+1],  G1 = h*phi2(Ah)B, G0 = h*(phi1-phi2)(Ah)B
//   y_{k+1} = Eyy y_k + c_k  -> chunked parallel scan (128 chunks x 64 steps)
//
// All matmuls run on v_wmma_f32_16x16x32_bf16 with a bf16 hi/lo split
// (3 WMMAs per 16x16x32 tile => ~fp32 accuracy, f32 accumulate).
// Operands are pre-split into bf16 hi/lo buffers so the WMMA inner loops are
// pure b128 loads + WMMA; ragged M is handled by buffer padding, not
// predication, so there is no exec-mask manipulation in the hot loops.
// ---------------------------------------------------------------------------

typedef __attribute__((ext_vector_type(16))) __bf16        bf16x16;
typedef __attribute__((ext_vector_type(8)))  unsigned int  u32x8;
typedef __attribute__((ext_vector_type(8)))  float         f32x8;
typedef unsigned short u16;

#define NDIM 1024
#define MDIM 512
#define TDIM 8192
#define LCHUNK 64
#define NCHUNK 128

static __device__ __forceinline__ u16 f2bf_rn(float f) {
  unsigned int u = __builtin_bit_cast(unsigned int, f);
  unsigned int lsb = (u >> 16) & 1u;
  u += 0x7fffu + lsb;
  return (u16)(u >> 16);
}
static __device__ __forceinline__ float bf2f(u16 h) {
  unsigned int u = ((unsigned int)h) << 16;
  return __builtin_bit_cast(float, u);
}

static __device__ __forceinline__ f32x8 zero_f32x8() {
  f32x8 z;
#pragma unroll
  for (int i = 0; i < 8; ++i) z[i] = 0.0f;
  return z;
}

// K base for VGPR slot v (holds K=kb,kb+1), per the ISA 16-bit A/B layout:
// half 0 (lanes 0-15): v0..3 -> K 0..7, v4..7 -> K 16..23
// half 1 (lanes16-31): v0..3 -> K 8..15, v4..7 -> K 24..31
static __device__ __forceinline__ int kmap(int v, int half) {
  return ((v < 4) ? 0 : 16) + (half ? 8 : 0) + ((v & 3) << 1);
}

// Load a 16x32 fragment from a pre-split bf16 buffer where the 32 K-values of
// row r are contiguous (row-major, ld elements): two 16B-contiguous dword
// quads -> the compiler merges them into global_load_b128 / ds_load_b128.
static __device__ __forceinline__ bf16x16 ld_frag(const u16* P, int ld, int r,
                                                  int k0, int half) {
  const unsigned int* p = (const unsigned int*)(P + (size_t)r * ld + k0);
  u32x8 d;
#pragma unroll
  for (int v = 0; v < 8; ++v) d[v] = p[kmap(v, half) >> 1];
  return __builtin_bit_cast(bf16x16, d);
}

// bf16x2-split product: hi*hi + hi*lo + lo*hi (lo*lo ~ 2^-32, dropped).
static __device__ __forceinline__ f32x8 wmma3(bf16x16 ah, bf16x16 al,
                                              bf16x16 bh, bf16x16 bl, f32x8 acc) {
  acc = __builtin_amdgcn_wmma_f32_16x16x32_bf16(false, ah, false, bh, (short)0, acc, false, false);
  acc = __builtin_amdgcn_wmma_f32_16x16x32_bf16(false, ah, false, bl, (short)0, acc, false, false);
  acc = __builtin_amdgcn_wmma_f32_16x16x32_bf16(false, al, false, bh, (short)0, acc, false, false);
  return acc;
}

// Split 8 consecutive-row f32 values into packed bf16 hi/lo dwords.
static __device__ __forceinline__ void split_pack8(const float* vals,
                                                   unsigned int* dhi, unsigned int* dlo) {
#pragma unroll
  for (int i = 0; i < 4; ++i) {
    u16 h0 = f2bf_rn(vals[2 * i]), h1 = f2bf_rn(vals[2 * i + 1]);
    u16 l0 = f2bf_rn(vals[2 * i] - bf2f(h0));
    u16 l1 = f2bf_rn(vals[2 * i + 1] - bf2f(h1));
    dhi[i] = (unsigned int)h0 | ((unsigned int)h1 << 16);
    dlo[i] = (unsigned int)l0 | ((unsigned int)l1 << 16);
  }
}

// ---------------------------------------------------------------------------
// GEMM on pre-split operands:
//   C = alpha * A @ B^T(col-major) + beta * C
// A: row-major M x K bf16 hi/lo (lda).  B: col-major [n][k] bf16 hi/lo (ldb=K).
// Optional epilogue outputs: row-major split (DhiR/DloR, ld=ldc) and
// col-major split (DhiC/DloC, [col][row], ld=M) of the f32 result.
// Block 256 threads (8 wave32) -> 128x64 tile; 4x2 waves, 32x32 per wave.
// Requires K % 32 == 0, N % 64 == 0, M % 128 == 0 (callers pad).
// ---------------------------------------------------------------------------
__global__ __launch_bounds__(256) void k_gemm_bs(
    const u16* __restrict__ Ahi, const u16* __restrict__ Alo, int lda,
    const u16* __restrict__ Bhi, const u16* __restrict__ Blo, int ldb,
    float* __restrict__ C, int ldc, int M, int N, int K, float alpha, float beta,
    u16* __restrict__ DhiR, u16* __restrict__ DloR,
    u16* __restrict__ DhiC, u16* __restrict__ DloC) {
  int lane = threadIdx.x & 31;
  int wave = threadIdx.x >> 5;
  int wm = wave & 3, wn = wave >> 2;
  int rowBase = blockIdx.y * 128 + wm * 32;
  int colBase = blockIdx.x * 64 + wn * 32;
  int half = (lane >> 4) & 1;
  int l15 = lane & 15;

  f32x8 acc[2][2];
#pragma unroll
  for (int i = 0; i < 2; ++i)
#pragma unroll
    for (int j = 0; j < 2; ++j) acc[i][j] = zero_f32x8();

  for (int k0 = 0; k0 < K; k0 += 32) {
    bf16x16 a0h = ld_frag(Ahi, lda, rowBase + l15, k0, half);
    bf16x16 a0l = ld_frag(Alo, lda, rowBase + l15, k0, half);
    bf16x16 a1h = ld_frag(Ahi, lda, rowBase + 16 + l15, k0, half);
    bf16x16 a1l = ld_frag(Alo, lda, rowBase + 16 + l15, k0, half);
    bf16x16 b0h = ld_frag(Bhi, ldb, colBase + l15, k0, half);
    bf16x16 b0l = ld_frag(Blo, ldb, colBase + l15, k0, half);
    bf16x16 b1h = ld_frag(Bhi, ldb, colBase + 16 + l15, k0, half);
    bf16x16 b1l = ld_frag(Blo, ldb, colBase + 16 + l15, k0, half);
    acc[0][0] = wmma3(a0h, a0l, b0h, b0l, acc[0][0]);
    acc[0][1] = wmma3(a0h, a0l, b1h, b1l, acc[0][1]);
    acc[1][0] = wmma3(a1h, a1l, b0h, b0l, acc[1][0]);
    acc[1][1] = wmma3(a1h, a1l, b1h, b1l, acc[1][1]);
  }

#pragma unroll
  for (int i = 0; i < 2; ++i)
#pragma unroll
    for (int j = 0; j < 2; ++j) {
#pragma unroll
      for (int v = 0; v < 8; ++v) {
        int row = rowBase + 16 * i + v + 8 * half;  // C/D layout: VGPR v -> M=v (+8 hi half)
        int col = colBase + 16 * j + l15;
        size_t idx = (size_t)row * ldc + col;
        float r = alpha * acc[i][j][v];
        if (beta != 0.0f) r += beta * C[idx];
        C[idx] = r;
        if (DhiR) {
          u16 hh = f2bf_rn(r);
          DhiR[idx] = hh;
          DloR[idx] = f2bf_rn(r - bf2f(hh));
        }
        if (DhiC) {
          size_t tidx = (size_t)col * M + row;
          u16 hh = f2bf_rn(r);
          DhiC[tidx] = hh;
          DloC[tidx] = f2bf_rn(r - bf2f(hh));
        }
      }
    }
}

// ---------------------------------------------------------------------------
// Chunked-scan pass 1 (local responses from zero state) and pass 3 (fix-up).
// 8 WGs x 1024 threads; each WG owns 16 time-chunks held as a split-bf16
// 1024x16 state in LDS (Whi/Wlo, 32 KB each).  Each of the 64 serial steps is
// Eyy(1024x1024) @ W(1024x16) via WMMA (32 waves x 2 row-tiles); state is
// re-split once per element per step at write time.
// ---------------------------------------------------------------------------
__global__ __launch_bounds__(1024) void k_scan_local(const u16* __restrict__ EyyHi,
                                                     const u16* __restrict__ EyyLo,
                                                     const float* __restrict__ Cm,
                                                     float* __restrict__ out) {
  __shared__ u16 Whi[16 * NDIM];
  __shared__ u16 Wlo[16 * NDIM];
  int tid = threadIdx.x, lane = tid & 31, wave = tid >> 5;
  int chunkBase = blockIdx.x * 16;
  for (int i = tid; i < 16 * NDIM; i += 1024) { Whi[i] = 0; Wlo[i] = 0; }
  __syncthreads();

  int half = (lane >> 4) & 1;
  int c = lane & 15;
  int m = chunkBase + c;
  int rowT = wave * 32;

  for (int step = 1; step <= LCHUNK; ++step) {
    f32x8 acc0 = zero_f32x8(), acc1 = zero_f32x8();
    for (int k0 = 0; k0 < NDIM; k0 += 32) {
      __builtin_prefetch(EyyHi + (size_t)(rowT + c) * NDIM + k0 + 128, 0, 1);
      bf16x16 a0h = ld_frag(EyyHi, NDIM, rowT + c, k0, half);
      bf16x16 a0l = ld_frag(EyyLo, NDIM, rowT + c, k0, half);
      bf16x16 a1h = ld_frag(EyyHi, NDIM, rowT + 16 + c, k0, half);
      bf16x16 a1l = ld_frag(EyyLo, NDIM, rowT + 16 + c, k0, half);
      bf16x16 bh = ld_frag(Whi, NDIM, c, k0, half);
      bf16x16 bl = ld_frag(Wlo, NDIM, c, k0, half);
      acc0 = wmma3(a0h, a0l, bh, bl, acc0);
      acc1 = wmma3(a1h, a1l, bh, bl, acc1);
    }
    __syncthreads();  // all reads of W complete before overwrite
    int t = m * LCHUNK + (step - 1);
    int krow = m * LCHUNK + step;
    int base0 = rowT + 8 * half;
    int base1 = rowT + 16 + 8 * half;
    float vals0[8], vals1[8];
#pragma unroll
    for (int v = 0; v < 8; ++v) {
      vals0[v] = acc0[v] + Cm[(size_t)t * NDIM + base0 + v];
      vals1[v] = acc1[v] + Cm[(size_t)t * NDIM + base1 + v];
    }
    split_pack8(vals0, (unsigned int*)(Whi + c * NDIM + base0),
                (unsigned int*)(Wlo + c * NDIM + base0));
    split_pack8(vals1, (unsigned int*)(Whi + c * NDIM + base1),
                (unsigned int*)(Wlo + c * NDIM + base1));
    if (krow < TDIM) {
      float* o = out + (size_t)krow * NDIM;
#pragma unroll
      for (int v = 0; v < 8; ++v) {
        o[base0 + v] = vals0[v];
        o[base1 + v] = vals1[v];
      }
    }
    __syncthreads();
  }
}

__global__ __launch_bounds__(1024) void k_scan_fixup(const u16* __restrict__ EyyHi,
                                                     const u16* __restrict__ EyyLo,
                                                     const float* __restrict__ Yb,
                                                     float* __restrict__ out) {
  __shared__ u16 Whi[16 * NDIM];
  __shared__ u16 Wlo[16 * NDIM];
  int tid = threadIdx.x, lane = tid & 31, wave = tid >> 5;
  int chunkBase = blockIdx.x * 16;
  for (int i = tid; i < 16 * NDIM; i += 1024) {  // Z_0 = boundary states
    int cc = i >> 10, rr = i & (NDIM - 1);
    float x = Yb[(size_t)(chunkBase + cc) * NDIM + rr];
    u16 hh = f2bf_rn(x);
    Whi[i] = hh;
    Wlo[i] = f2bf_rn(x - bf2f(hh));
  }
  __syncthreads();

  int half = (lane >> 4) & 1;
  int c = lane & 15;
  int m = chunkBase + c;
  int rowT = wave * 32;

  for (int step = 1; step <= LCHUNK; ++step) {
    f32x8 acc0 = zero_f32x8(), acc1 = zero_f32x8();
    for (int k0 = 0; k0 < NDIM; k0 += 32) {
      bf16x16 a0h = ld_frag(EyyHi, NDIM, rowT + c, k0, half);
      bf16x16 a0l = ld_frag(EyyLo, NDIM, rowT + c, k0, half);
      bf16x16 a1h = ld_frag(EyyHi, NDIM, rowT + 16 + c, k0, half);
      bf16x16 a1l = ld_frag(EyyLo, NDIM, rowT + 16 + c, k0, half);
      bf16x16 bh = ld_frag(Whi, NDIM, c, k0, half);
      bf16x16 bl = ld_frag(Wlo, NDIM, c, k0, half);
      acc0 = wmma3(a0h, a0l, bh, bl, acc0);
      acc1 = wmma3(a1h, a1l, bh, bl, acc1);
    }
    __syncthreads();
    int krow = m * LCHUNK + step;
    int base0 = rowT + 8 * half;
    int base1 = rowT + 16 + 8 * half;
    float vals0[8], vals1[8];
#pragma unroll
    for (int v = 0; v < 8; ++v) { vals0[v] = acc0[v]; vals1[v] = acc1[v]; }
    split_pack8(vals0, (unsigned int*)(Whi + c * NDIM + base0),
                (unsigned int*)(Wlo + c * NDIM + base0));
    split_pack8(vals1, (unsigned int*)(Whi + c * NDIM + base1),
                (unsigned int*)(Wlo + c * NDIM + base1));
    if (krow < TDIM) {
      float* o = out + (size_t)krow * NDIM;
#pragma unroll
      for (int v = 0; v < 8; ++v) {
        o[base0 + v] += vals0[v];
        o[base1 + v] += vals1[v];
      }
    }
    __syncthreads();
  }
}

// Boundary recurrence: Yb[0]=y0; Yb[m]=Eyy^L Yb[m-1] + d_{m-1},
// d_{m-1} = local response at previous chunk end = out row m*L (pass-1 value).
__global__ __launch_bounds__(1024) void k_scan_boundary(const float* __restrict__ EyyL,
                                                        const float* __restrict__ y0,
                                                        const float* __restrict__ out,
                                                        float* __restrict__ Yb) {
  __shared__ float y[NDIM];
  int tid = threadIdx.x;
  float v = y0[tid];
  y[tid] = v;
  Yb[tid] = v;
  __syncthreads();
  for (int mm = 1; mm < NCHUNK; ++mm) {
    const float* row = EyyL + (size_t)tid * NDIM;
    float s = 0.0f;
    for (int k = 0; k < NDIM; ++k) s = fmaf(row[k], y[k], s);
    s += out[(size_t)(mm * LCHUNK) * NDIM + tid];
    __syncthreads();
    y[tid] = s;
    Yb[(size_t)mm * NDIM + tid] = s;
    __syncthreads();
  }
}

// -------------------------- small utility kernels --------------------------
__global__ void k_scale(float* __restrict__ d, const float* __restrict__ s, float a, int n) {
  int i = blockIdx.x * blockDim.x + threadIdx.x;
  if (i < n) d[i] = a * s[i];
}
__global__ void k_axpy(float* __restrict__ d, const float* __restrict__ s, float a, int n) {
  int i = blockIdx.x * blockDim.x + threadIdx.x;
  if (i < n) d[i] += a * s[i];
}
__global__ void k_addI(float* __restrict__ m, int n, float a) {
  int i = blockIdx.x * blockDim.x + threadIdx.x;
  if (i < n) m[(size_t)i * n + i] += a;
}
__global__ void k_copy(float* __restrict__ d, const float* __restrict__ s, int n) {
  int i = blockIdx.x * blockDim.x + threadIdx.x;
  if (i < n) d[i] = s[i];
}
__global__ void k_zero(float* __restrict__ d, int n) {
  int i = blockIdx.x * blockDim.x + threadIdx.x;
  if (i < n) d[i] = 0.0f;
}
__global__ void k_zero_u16(u16* __restrict__ d, int n) {
  int i = blockIdx.x * blockDim.x + threadIdx.x;
  if (i < n) d[i] = 0;
}
// Split f32 -> bf16 hi/lo, same (row-major) layout.
__global__ void k_rsplit(u16* __restrict__ hi, u16* __restrict__ lo,
                         const float* __restrict__ s, int n) {
  int i = blockIdx.x * blockDim.x + threadIdx.x;
  if (i < n) {
    float x = s[i];
    u16 h = f2bf_rn(x);
    hi[i] = h;
    lo[i] = f2bf_rn(x - bf2f(h));
  }
}
// Split f32 (row-major rows x cols) -> bf16 hi/lo transposed: out[c*rows + r].
__global__ void k_csplit(u16* __restrict__ hi, u16* __restrict__ lo,
                         const float* __restrict__ s, int rows, int cols) {
  int i = blockIdx.x * blockDim.x + threadIdx.x;
  if (i < rows * cols) {
    int r = i / cols, c = i - r * cols;
    float x = s[i];
    u16 h = f2bf_rn(x);
    size_t o = (size_t)c * rows + r;
    hi[o] = h;
    lo[o] = f2bf_rn(x - bf2f(h));
  }
}

// ---------------------------------------------------------------------------
extern "C" void kernel_launch(void* const* d_in, const int* in_sizes, int n_in,
                              void* d_out, int out_size, void* d_ws, size_t ws_size,
                              hipStream_t stream) {
  (void)in_sizes; (void)n_in; (void)out_size; (void)ws_size;
  const float* y0v = (const float*)d_in[1];
  const float* u   = (const float*)d_in[2];
  const float* Am  = (const float*)d_in[3];
  const float* Bm  = (const float*)d_in[4];
  float* outF = (float*)d_out;

  const float h = 0.01f;
  const int NN = NDIM * NDIM;
  const int NM = NDIM * MDIM;

  // ---- workspace layout ----
  char* W = (char*)d_ws;
  size_t off = 0;
  auto alloc = [&](size_t bytes) { void* p = W + off; off += (bytes + 255) & ~(size_t)255; return p; };
  // persistent through the scan
  u16*   EyyHi = (u16*)alloc(NN * 2);          // e^{Ah}, split, row-major
  u16*   EyyLo = (u16*)alloc(NN * 2);
  float* EyyL  = (float*)alloc(NN * 4);        // Eyy^64 (f32, boundary pass)
  float* Yb    = (float*)alloc((size_t)NCHUNK * NDIM * 4);
  float* Cm    = (float*)alloc((size_t)TDIM * NDIM * 4);
  u16*   G0Hi  = (u16*)alloc(NM * 2);          // G split, row-major [n][k], ld=512
  u16*   G0Lo  = (u16*)alloc(NM * 2);
  u16*   G1Hi  = (u16*)alloc(NM * 2);
  u16*   G1Lo  = (u16*)alloc(NM * 2);
  // phase scratch
  size_t scratch0 = off;
  float* Ah     = (float*)alloc(NN * 4);
  u16*   AhRsHi = (u16*)alloc(NN * 2);
  u16*   AhRsLo = (u16*)alloc(NN * 2);
  u16*   AhCsHi = (u16*)alloc(NN * 2);
  u16*   AhCsLo = (u16*)alloc(NN * 2);
  float* Nxt    = (float*)alloc(NN * 4);
  u16*   MkRsHi0 = (u16*)alloc(NN * 2);
  u16*   MkRsLo0 = (u16*)alloc(NN * 2);
  u16*   MkRsHi1 = (u16*)alloc(NN * 2);
  u16*   MkRsLo1 = (u16*)alloc(NN * 2);
  u16*   MkCsHi0 = (u16*)alloc(NN * 2);
  u16*   MkCsLo0 = (u16*)alloc(NN * 2);
  u16*   MkCsHi1 = (u16*)alloc(NN * 2);
  u16*   MkCsLo1 = (u16*)alloc(NN * 2);
  float* Eyy    = (float*)alloc(NN * 4);
  u16*   EyyCsHi = (u16*)alloc(NN * 2);
  u16*   EyyCsLo = (u16*)alloc(NN * 2);
  float* S1     = (float*)alloc(NN * 4);
  float* S2     = (float*)alloc(NN * 4);
  u16*   SRsHi  = (u16*)alloc(NN * 2);
  u16*   SRsLo  = (u16*)alloc(NN * 2);
  float* Gf     = (float*)alloc(NM * 4);
  u16*   BmCsHi = (u16*)alloc(NM * 2);
  u16*   BmCsLo = (u16*)alloc(NM * 2);
  // phase-B alias: u's split (with one zero padding row so the second C-GEMM
  // can run guard-free at M=8192) reuses the dead head of the scratch area.
  u16* uHi = (u16*)(W + scratch0);
  u16* uLo = uHi + (size_t)(TDIM + 1) * MDIM;

  dim3 b256(256);
  int gNN = (NN + 255) / 256;
  int gNM = (NM + 255) / 256;

  // --- Taylor series: Eyy=phi0, S1=phi1, S2=phi2 sums (f32) ---
  k_scale<<<gNN, b256, 0, stream>>>(Ah, Am, h, NN);
  k_rsplit<<<gNN, b256, 0, stream>>>(AhRsHi, AhRsLo, Ah, NN);
  k_csplit<<<gNN, b256, 0, stream>>>(AhCsHi, AhCsLo, Ah, NDIM, NDIM);
  k_scale<<<gNN, b256, 0, stream>>>(Eyy, Ah, 1.0f, NN);
  k_addI<<<4, b256, 0, stream>>>(Eyy, NDIM, 1.0f);
  k_scale<<<gNN, b256, 0, stream>>>(S1, Ah, 0.5f, NN);
  k_addI<<<4, b256, 0, stream>>>(S1, NDIM, 1.0f);
  k_scale<<<gNN, b256, 0, stream>>>(S2, Ah, 1.0f / 6.0f, NN);
  k_addI<<<4, b256, 0, stream>>>(S2, NDIM, 0.5f);

  dim3 gemmSq(NDIM / 64, NDIM / 128);  // 16 x 8
  const u16* curHi = AhRsHi;
  const u16* curLo = AhRsLo;
  int ping = 0;
  double fact = 1.0;
  for (int k = 2; k <= 12; ++k) {
    u16* dHi = ping ? MkRsHi1 : MkRsHi0;
    u16* dLo = ping ? MkRsLo1 : MkRsLo0;
    k_gemm_bs<<<gemmSq, b256, 0, stream>>>(curHi, curLo, NDIM, AhCsHi, AhCsLo, NDIM,
                                           Nxt, NDIM, NDIM, NDIM, NDIM, 1.0f, 0.0f,
                                           dHi, dLo, nullptr, nullptr);
    fact *= (double)k;
    float c0 = (float)(1.0 / fact);
    float c1 = (float)(1.0 / (fact * (double)(k + 1)));
    float c2 = (float)(1.0 / (fact * (double)(k + 1) * (double)(k + 2)));
    k_axpy<<<gNN, b256, 0, stream>>>(Eyy, Nxt, c0, NN);
    k_axpy<<<gNN, b256, 0, stream>>>(S1, Nxt, c1, NN);
    k_axpy<<<gNN, b256, 0, stream>>>(S2, Nxt, c2, NN);
    curHi = dHi; curLo = dLo; ping ^= 1;
  }

  // --- persistent splits of Eyy; EyyL = Eyy^64 via 6 squarings ---
  k_rsplit<<<gNN, b256, 0, stream>>>(EyyHi, EyyLo, Eyy, NN);
  k_csplit<<<gNN, b256, 0, stream>>>(EyyCsHi, EyyCsLo, Eyy, NDIM, NDIM);
  const u16 *cRsH = EyyHi, *cRsL = EyyLo, *cCsH = EyyCsHi, *cCsL = EyyCsLo;
  ping = 0;
  for (int sq = 0; sq < 6; ++sq) {
    bool last = (sq == 5);
    float* cf = last ? EyyL : Nxt;
    u16* dRsH = last ? nullptr : (ping ? MkRsHi1 : MkRsHi0);
    u16* dRsL = last ? nullptr : (ping ? MkRsLo1 : MkRsLo0);
    u16* dCsH = last ? nullptr : (ping ? MkCsHi1 : MkCsHi0);
    u16* dCsL = last ? nullptr : (ping ? MkCsLo1 : MkCsLo0);
    k_gemm_bs<<<gemmSq, b256, 0, stream>>>(cRsH, cRsL, NDIM, cCsH, cCsL, NDIM,
                                           cf, NDIM, NDIM, NDIM, NDIM, 1.0f, 0.0f,
                                           dRsH, dRsL, dCsH, dCsL);
    if (!last) { cRsH = dRsH; cRsL = dRsL; cCsH = dCsH; cCsL = dCsL; ping ^= 1; }
  }

  // --- G1 = h*phi2@B ; G0 = h*(phi1-phi2)@B  (split outputs for the C GEMM) ---
  dim3 gemmNB(MDIM / 64, NDIM / 128);  // 8 x 8
  k_csplit<<<gNM, b256, 0, stream>>>(BmCsHi, BmCsLo, Bm, NDIM, MDIM);
  k_rsplit<<<gNN, b256, 0, stream>>>(SRsHi, SRsLo, S2, NN);
  k_gemm_bs<<<gemmNB, b256, 0, stream>>>(SRsHi, SRsLo, NDIM, BmCsHi, BmCsLo, NDIM,
                                         Gf, MDIM, NDIM, MDIM, NDIM, h, 0.0f,
                                         G1Hi, G1Lo, nullptr, nullptr);
  k_axpy<<<gNN, b256, 0, stream>>>(S1, S2, -1.0f, NN);  // S1 := phi1 - phi2
  k_rsplit<<<gNN, b256, 0, stream>>>(SRsHi, SRsLo, S1, NN);
  k_gemm_bs<<<gemmNB, b256, 0, stream>>>(SRsHi, SRsLo, NDIM, BmCsHi, BmCsLo, NDIM,
                                         Gf, MDIM, NDIM, MDIM, NDIM, h, 0.0f,
                                         G0Hi, G0Lo, nullptr, nullptr);

  // --- C[t] = u[t]@G0^T + u[t+1]@G1^T, t = 0..T-2; row T-1 zeroed.
  // u split is padded with one zero row; both GEMMs run at M=8192 (tile-exact,
  // no guards).  The garbage accumulated into Cm row 8191 is zeroed after. ---
  int gU = ((int)((size_t)TDIM * MDIM) + 255) / 256;
  k_rsplit<<<gU, b256, 0, stream>>>(uHi, uLo, u, TDIM * MDIM);
  k_zero_u16<<<2, b256, 0, stream>>>(uHi + (size_t)TDIM * MDIM, MDIM);
  k_zero_u16<<<2, b256, 0, stream>>>(uLo + (size_t)TDIM * MDIM, MDIM);
  dim3 gemmC(NDIM / 64, TDIM / 128);  // 16 x 64
  k_gemm_bs<<<gemmC, b256, 0, stream>>>(uHi, uLo, MDIM, G0Hi, G0Lo, MDIM,
                                        Cm, NDIM, TDIM, NDIM, MDIM, 1.0f, 0.0f,
                                        nullptr, nullptr, nullptr, nullptr);
  k_gemm_bs<<<gemmC, b256, 0, stream>>>(uHi + MDIM, uLo + MDIM, MDIM, G1Hi, G1Lo, MDIM,
                                        Cm, NDIM, TDIM, NDIM, MDIM, 1.0f, 1.0f,
                                        nullptr, nullptr, nullptr, nullptr);
  k_zero<<<4, b256, 0, stream>>>(Cm + (size_t)(TDIM - 1) * NDIM, NDIM);

  // --- output row 0 = y0 ---
  k_copy<<<4, b256, 0, stream>>>(outF, y0v, NDIM);

  // --- chunked scan: local pass, boundary recurrence, fix-up ---
  k_scan_local<<<NCHUNK / 16, 1024, 0, stream>>>(EyyHi, EyyLo, Cm, outF);
  k_scan_boundary<<<1, 1024, 0, stream>>>(EyyL, y0v, outF, Yb);
  k_scan_fixup<<<NCHUNK / 16, 1024, 0, stream>>>(EyyHi, EyyLo, Yb, outF);
}
